// MultiHeadAttention_6116033429738
// MI455X (gfx1250) — matrix operand order repaired and checked by reading, not compiled, not run on hardware
//
#include <hip/hip_runtime.h>

#define DIM   1024
#define NH    16
#define HD    64
#define BB    2
#define SS    2048
#define NROW  (BB * SS)      // 4096 rows of x
#define N3    (3 * DIM)      // 3072
#define ATT_SCALE 0.125f     // 1/sqrt(64)

#define PAD32 40             // padded LDS row (elems) for 32-wide tiles, 80B (bank-safe, 16B aligned)
#define PAD64 72             // padded LDS row (elems) for 64-wide tiles, 144B

typedef __attribute__((ext_vector_type(16))) __bf16          v16bf;
typedef __attribute__((ext_vector_type(8)))  float           v8f;
typedef __attribute__((ext_vector_type(8)))  unsigned short  v8u16;
typedef __attribute__((ext_vector_type(4)))  int             v4i;

struct Frag { v8u16 lo, hi; };

#if defined(__AMDGCN__) && __has_builtin(__builtin_amdgcn_global_load_async_to_lds_b128)
#define HAS_ASYNC_LDS 1
#else
#define HAS_ASYNC_LDS 0
#endif

typedef __attribute__((address_space(1))) v4i *g128p;   // global int4*
typedef __attribute__((address_space(3))) v4i *l128p;   // LDS int4*

// 16-byte global -> LDS copy: async (ASYNCcnt) on CDNA5, sync fallback otherwise.
__device__ __forceinline__ void cp16(unsigned short* l, const unsigned short* g) {
#if HAS_ASYNC_LDS
  __builtin_amdgcn_global_load_async_to_lds_b128((g128p)(v4i*)g, (l128p)(v4i*)l, 0, 0);
#else
  *(v8u16*)l = *(const v8u16*)g;
#endif
}
__device__ __forceinline__ void cp_wait_all() {
#if HAS_ASYNC_LDS
  asm volatile("s_wait_asynccnt 0" ::: "memory");
#endif
}

__device__ __forceinline__ unsigned short f2bf(float f) {
  unsigned int u = __float_as_uint(f);
  u += 0x7fffu + ((u >> 16) & 1u);   // round-to-nearest-even
  return (unsigned short)(u >> 16);
}

__device__ __forceinline__ v8f wmma_bf16(v16bf a, v16bf b, v8f c) {
  return __builtin_amdgcn_wmma_f32_16x16x32_bf16(false, a, false, b, (short)0, c,
                                                 false, false);
}

// A fragment (16x32 bf16, row-major source, works for global or LDS pointers).
// Lane l: row = l&15; K blocks {0..7,16..23} (l<16) / {8..15,24..31} (l>=16).
__device__ __forceinline__ v16bf load_a_frag(const unsigned short* A, int lda,
                                             int row0, int k0) {
  int lane = threadIdx.x & 31;
  int m = lane & 15, hi = lane >> 4;
  const unsigned short* p = A + (size_t)(row0 + m) * lda + (size_t)(k0 + hi * 8);
  Frag f;
  f.lo = *(const v8u16*)p;
  f.hi = *(const v8u16*)(p + 16);
  return __builtin_bit_cast(v16bf, f);
}

// B fragment for C = A * B^T with B stored row-major [N][K].
// Lane l: column l&15; K blocks {0..7,16..23} (l<16) / {8..15,24..31} (l>=16).
__device__ __forceinline__ v16bf load_bt_frag(const unsigned short* B, int ldb,
                                              int col0, int k0) {
  int lane = threadIdx.x & 31;
  int n = lane & 15, hi = lane >> 4;
  const unsigned short* p = B + (size_t)(col0 + n) * ldb + (size_t)(k0 + hi * 8);
  Frag f;
  f.lo = *(const v8u16*)p;
  f.hi = *(const v8u16*)(p + 16);
  return __builtin_bit_cast(v16bf, f);
}

// ---------------- fp32 -> bf16 convert (x4 vectorized) ----------------
__global__ void cvt_bf16_kernel(const float* __restrict__ in,
                                unsigned short* __restrict__ out, int n4) {
  int i = blockIdx.x * blockDim.x + threadIdx.x;
  if (i < n4) {
    float4 v = ((const float4*)in)[i];
    ushort4 o;
    o.x = f2bf(v.x); o.y = f2bf(v.y); o.z = f2bf(v.z); o.w = f2bf(v.w);
    ((ushort4*)out)[i] = o;
  }
}

// ---------------- QKV projection: qkv = x @ qkv_w^T + b ----------------
// Block tile 128x128, wave tile 32x64 (8 WMMAs per k-step), async
// double-buffered LDS staging of A and B, WMMA operands from LDS.
__global__ __launch_bounds__(256) void qkv_gemm_kernel(
    const unsigned short* __restrict__ xbf, const unsigned short* __restrict__ wbf,
    const float* __restrict__ bias, unsigned short* __restrict__ qbf,
    unsigned short* __restrict__ kbf, unsigned short* __restrict__ vtbf) {
  __shared__ __align__(16) unsigned short At[2][128][PAD32];
  __shared__ __align__(16) unsigned short Bt[2][128][PAD32];

  const int t  = threadIdx.x;
  const int w  = t >> 5;
  const int mw = (w & 3) * 32;           // wave M sub-tile (32 rows)
  const int nw = (w >> 2) * 64;          // wave N sub-tile (4 frags of 16)
  const int rowBase = blockIdx.x * 128;
  const int colBase = blockIdx.y * 128;

  // copy assignments: 16B pieces; each tile 128x32 = 512 pieces -> 2 rows/thread
  const int cr = t >> 2, co = (t & 3) * 8;

  cp16(&At[0][cr][co],      xbf + (size_t)(rowBase + cr) * DIM + co);
  cp16(&At[0][cr + 64][co], xbf + (size_t)(rowBase + cr + 64) * DIM + co);
  cp16(&Bt[0][cr][co],      wbf + (size_t)(colBase + cr) * DIM + co);
  cp16(&Bt[0][cr + 64][co], wbf + (size_t)(colBase + cr + 64) * DIM + co);

  v8f acc[2][4] = {};
#pragma unroll 1   // keep buf dynamic: accumulators stay pinned, no inter-body reg shuffling
  for (int ks = 0; ks < DIM / 32; ++ks) {
    cp_wait_all();
    __syncthreads();
    const int buf = ks & 1;
    if (ks + 1 < DIM / 32) {
      const int kn = (ks + 1) * 32;
      cp16(&At[buf ^ 1][cr][co],      xbf + (size_t)(rowBase + cr) * DIM + kn + co);
      cp16(&At[buf ^ 1][cr + 64][co], xbf + (size_t)(rowBase + cr + 64) * DIM + kn + co);
      cp16(&Bt[buf ^ 1][cr][co],      wbf + (size_t)(colBase + cr) * DIM + kn + co);
      cp16(&Bt[buf ^ 1][cr + 64][co], wbf + (size_t)(colBase + cr + 64) * DIM + kn + co);
    }
    v16bf a0 = load_a_frag(&At[buf][0][0], PAD32, mw, 0);
    v16bf a1 = load_a_frag(&At[buf][0][0], PAD32, mw + 16, 0);
#pragma unroll
    for (int nf = 0; nf < 4; ++nf) {
      v16bf b = load_bt_frag(&Bt[buf][0][0], PAD32, nw + nf * 16, 0);
      acc[0][nf] = wmma_bf16(a0, b, acc[0][nf]);
      acc[1][nf] = wmma_bf16(a1, b, acc[1][nf]);
    }
  }

  const int lane = t & 31;
  const int cn = lane & 15, hi = lane >> 4;
#pragma unroll
  for (int mi = 0; mi < 2; ++mi) {
#pragma unroll
    for (int nf = 0; nf < 4; ++nf) {
      int n = colBase + nw + nf * 16 + cn;
      float bv  = bias[n];
      int which = n >> 10;           // 0=q 1=k 2=v (uniform per fragment)
      int rem   = n & 1023;
      int h     = rem >> 6;
      int d     = rem & 63;
#pragma unroll
      for (int r = 0; r < 8; ++r) {
        int row = rowBase + mw + mi * 16 + r + hi * 8;   // b*SS + s
        int b = row >> 11, s = row & (SS - 1);
        unsigned short val = f2bf(acc[mi][nf][r] + bv);
        size_t bh = (size_t)(b * NH + h);
        if (which == 0)      qbf [(bh * SS + s) * HD + d] = val;
        else if (which == 1) kbf [(bh * SS + s) * HD + d] = val;
        else                 vtbf[(bh * HD + d) * SS + s] = val;
      }
    }
  }
}

// ---------------- Flash attention (online softmax) ----------------
// Grid: (B*H, S/128). 8 waves/block each own 16 query rows; the block
// cooperatively streams K/V^T 32-key chunks through double-buffered LDS
// via async copies; softmax VALU work overlaps the next chunk's DMA.
__global__ __launch_bounds__(256) void flash_attn_kernel(
    const unsigned short* __restrict__ qbf, const unsigned short* __restrict__ kbf,
    const unsigned short* __restrict__ vtbf, unsigned short* __restrict__ obf) {
  __shared__ __align__(16) unsigned short Kt[2][32][PAD64];   // 32 keys x 64 d
  __shared__ __align__(16) unsigned short Vt[2][64][PAD32];   // 64 d x 32 keys
  __shared__ __align__(16) unsigned short Pt[8][16][PAD32];   // per-wave P tile

  const int t = threadIdx.x, w = t >> 5, lane = t & 31;
  const int cn = lane & 15, hi = lane >> 4;
  const int bh = blockIdx.x;
  const int q0 = blockIdx.y * 128 + w * 16;

  const unsigned short* Q  = qbf  + (size_t)bh * SS * HD;
  const unsigned short* K  = kbf  + (size_t)bh * SS * HD;
  const unsigned short* VT = vtbf + (size_t)bh * HD * SS;

  // copy assignments: K chunk 32x64 (8 pieces/row), V^T chunk 64x32 (4 pieces/row)
  const int kr = t >> 3, ko = (t & 7) * 8;
  const int vr = t >> 2, vo = (t & 3) * 8;

  cp16(&Kt[0][kr][ko], K  + (size_t)kr * HD + ko);
  cp16(&Vt[0][vr][vo], VT + (size_t)vr * SS + vo);

  // Q tile 16x64 resident in registers (2 A-fragments).
  v16bf qf0 = load_a_frag(Q, HD, q0, 0);
  v16bf qf1 = load_a_frag(Q, HD, q0, 32);

  v8f acco[4] = {};
  float mrun[8], lrun[8];
#pragma unroll
  for (int r = 0; r < 8; ++r) { mrun[r] = -1e30f; lrun[r] = 0.0f; }

#pragma unroll 1   // dynamic buf: keep acco/stat registers pinned across chunks
  for (int c = 0; c < SS / 32; ++c) {
    cp_wait_all();
    __syncthreads();
    const int buf = c & 1;
    if (c + 1 < SS / 32) {
      const int j0n = (c + 1) * 32;
      cp16(&Kt[buf ^ 1][kr][ko], K  + (size_t)(j0n + kr) * HD + ko);
      cp16(&Vt[buf ^ 1][vr][vo], VT + (size_t)vr * SS + j0n + vo);
    }

    // scores S = Q K^T (16x32), f32 accum, operands from LDS
    v8f s0 = {}, s1 = {};
    s0 = wmma_bf16(qf0, load_bt_frag(&Kt[buf][0][0], PAD64, 0,  0),  s0);
    s0 = wmma_bf16(qf1, load_bt_frag(&Kt[buf][0][0], PAD64, 0,  32), s0);
    s1 = wmma_bf16(qf0, load_bt_frag(&Kt[buf][0][0], PAD64, 16, 0),  s1);
    s1 = wmma_bf16(qf1, load_bt_frag(&Kt[buf][0][0], PAD64, 16, 32), s1);

    float alpha[8];
#pragma unroll
    for (int r = 0; r < 8; ++r) {
      float a = s0[r] * ATT_SCALE;
      float b = s1[r] * ATT_SCALE;
      float mx = fmaxf(a, b);                 // row max across 16-lane half
      mx = fmaxf(mx, __shfl_xor(mx, 1, 32));
      mx = fmaxf(mx, __shfl_xor(mx, 2, 32));
      mx = fmaxf(mx, __shfl_xor(mx, 4, 32));
      mx = fmaxf(mx, __shfl_xor(mx, 8, 32));
      float mnew = fmaxf(mrun[r], mx);
      alpha[r] = __expf(mrun[r] - mnew);
      float p0 = __expf(a - mnew);
      float p1 = __expf(b - mnew);
      s0[r] = p0; s1[r] = p1;
      float u = p0 + p1;                      // row sum
      u += __shfl_xor(u, 1, 32);
      u += __shfl_xor(u, 2, 32);
      u += __shfl_xor(u, 4, 32);
      u += __shfl_xor(u, 8, 32);
      lrun[r] = lrun[r] * alpha[r] + u;
      mrun[r] = mnew;
    }
#pragma unroll
    for (int f = 0; f < 4; ++f)
#pragma unroll
      for (int r = 0; r < 8; ++r) acco[f][r] *= alpha[r];

    // stage P (16x32) through this wave's private LDS tile -> A layout
#pragma unroll
    for (int r = 0; r < 8; ++r) {
      Pt[w][r + hi * 8][cn]      = f2bf(s0[r]);
      Pt[w][r + hi * 8][16 + cn] = f2bf(s1[r]);
    }
    asm volatile("s_wait_dscnt 0" ::: "memory");
    v16bf pf = load_a_frag(&Pt[w][0][0], PAD32, 0, 0);

    // O += P * V  (V^T chunk in LDS, contiguous along keys)
#pragma unroll
    for (int f = 0; f < 4; ++f) {
      v16bf vf = load_bt_frag(&Vt[buf][0][0], PAD32, f * 16, 0);
      acco[f] = wmma_bf16(pf, vf, acco[f]);
    }
  }

  // normalize and emit bf16 attention output, row-major (B,S,DIM)
  const int b = bh >> 4, h = bh & (NH - 1);
  float inv[8];
#pragma unroll
  for (int r = 0; r < 8; ++r) inv[r] = 1.0f / lrun[r];
#pragma unroll
  for (int f = 0; f < 4; ++f) {
    int d = f * 16 + cn;
#pragma unroll
    for (int r = 0; r < 8; ++r) {
      int s = q0 + r + hi * 8;
      obf[((size_t)(b * SS + s)) * DIM + h * HD + d] = f2bf(acco[f][r] * inv[r]);
    }
  }
}

// ---------------- Output projection: out = attn @ proj_w^T + b (fp32) ----------------
__global__ __launch_bounds__(256) void proj_gemm_kernel(
    const unsigned short* __restrict__ abf, const unsigned short* __restrict__ wbf,
    const float* __restrict__ bias, float* __restrict__ out) {
  __shared__ __align__(16) unsigned short At[2][128][PAD32];
  __shared__ __align__(16) unsigned short Bt[2][128][PAD32];

  const int t  = threadIdx.x;
  const int w  = t >> 5;
  const int mw = (w & 3) * 32;
  const int nw = (w >> 2) * 64;
  const int rowBase = blockIdx.x * 128;
  const int colBase = blockIdx.y * 128;
  const int cr = t >> 2, co = (t & 3) * 8;

  cp16(&At[0][cr][co],      abf + (size_t)(rowBase + cr) * DIM + co);
  cp16(&At[0][cr + 64][co], abf + (size_t)(rowBase + cr + 64) * DIM + co);
  cp16(&Bt[0][cr][co],      wbf + (size_t)(colBase + cr) * DIM + co);
  cp16(&Bt[0][cr + 64][co], wbf + (size_t)(colBase + cr + 64) * DIM + co);

  v8f acc[2][4] = {};
#pragma unroll 1   // keep buf dynamic: accumulators stay pinned, no inter-body reg shuffling
  for (int ks = 0; ks < DIM / 32; ++ks) {
    cp_wait_all();
    __syncthreads();
    const int buf = ks & 1;
    if (ks + 1 < DIM / 32) {
      const int kn = (ks + 1) * 32;
      cp16(&At[buf ^ 1][cr][co],      abf + (size_t)(rowBase + cr) * DIM + kn + co);
      cp16(&At[buf ^ 1][cr + 64][co], abf + (size_t)(rowBase + cr + 64) * DIM + kn + co);
      cp16(&Bt[buf ^ 1][cr][co],      wbf + (size_t)(colBase + cr) * DIM + kn + co);
      cp16(&Bt[buf ^ 1][cr + 64][co], wbf + (size_t)(colBase + cr + 64) * DIM + kn + co);
    }
    v16bf a0 = load_a_frag(&At[buf][0][0], PAD32, mw, 0);
    v16bf a1 = load_a_frag(&At[buf][0][0], PAD32, mw + 16, 0);
#pragma unroll
    for (int nf = 0; nf < 4; ++nf) {
      v16bf b = load_bt_frag(&Bt[buf][0][0], PAD32, nw + nf * 16, 0);
      acc[0][nf] = wmma_bf16(a0, b, acc[0][nf]);
      acc[1][nf] = wmma_bf16(a1, b, acc[1][nf]);
    }
  }

  const int lane = t & 31;
  const int cn = lane & 15, hi = lane >> 4;
#pragma unroll
  for (int mi = 0; mi < 2; ++mi) {
#pragma unroll
    for (int nf = 0; nf < 4; ++nf) {
      int n = colBase + nw + nf * 16 + cn;
      float bv = bias[n];
#pragma unroll
      for (int r = 0; r < 8; ++r) {
        int row = rowBase + mw + mi * 16 + r + hi * 8;
        out[(size_t)row * DIM + n] = acc[mi][nf][r] + bv;
      }
    }
  }
}

extern "C" void kernel_launch(void* const* d_in, const int* in_sizes, int n_in,
                              void* d_out, int out_size, void* d_ws, size_t ws_size,
                              hipStream_t stream) {
  const float* x      = (const float*)d_in[0];
  const float* qkv_w  = (const float*)d_in[1];
  const float* qkv_b  = (const float*)d_in[2];
  const float* proj_w = (const float*)d_in[3];
  const float* proj_b = (const float*)d_in[4];
  float* out = (float*)d_out;

  // workspace layout (bf16 = ushort), ~48 MiB total
  unsigned short* xbf     = (unsigned short*)d_ws;
  unsigned short* qkvwbf  = xbf     + (size_t)NROW * DIM;
  unsigned short* projwbf = qkvwbf  + (size_t)N3   * DIM;
  unsigned short* qbf     = projwbf + (size_t)DIM  * DIM;
  unsigned short* kbf     = qbf     + (size_t)BB * NH * SS * HD;
  unsigned short* vtbf    = kbf     + (size_t)BB * NH * SS * HD;
  unsigned short* obf     = vtbf    + (size_t)BB * NH * SS * HD;

  int n4;
  n4 = (NROW * DIM) / 4;
  cvt_bf16_kernel<<<n4 / 256, 256, 0, stream>>>(x, xbf, n4);
  n4 = (N3 * DIM) / 4;
  cvt_bf16_kernel<<<n4 / 256, 256, 0, stream>>>(qkv_w, qkvwbf, n4);
  n4 = (DIM * DIM) / 4;
  cvt_bf16_kernel<<<n4 / 256, 256, 0, stream>>>(proj_w, projwbf, n4);

  qkv_gemm_kernel<<<dim3(NROW / 128, N3 / 128), 256, 0, stream>>>(
      xbf, qkvwbf, qkv_b, qbf, kbf, vtbf);

  flash_attn_kernel<<<dim3(BB * NH, SS / 128), 256, 0, stream>>>(
      qbf, kbf, vtbf, obf);

  proj_gemm_kernel<<<dim3(NROW / 128, DIM / 128), 256, 0, stream>>>(
      obf, projwbf, proj_b, out);
}
